// STCIMwithGCN_48455821033623
// MI455X (gfx1250) — matrix operand order repaired and checked by reading, not compiled, not run on hardware
//
#include <hip/hip_runtime.h>
#include <hip/hip_bf16.h>

typedef __attribute__((ext_vector_type(16))) __bf16 v16bf;
typedef __attribute__((ext_vector_type(8)))  float  v8f;

#define NB 32
#define NN 1024
#define CCH 32
#define SKIPCH 64
#define NDIM 40
#define KPAD 64
#define NNZ 24
#define COLS (NB*NN)   // 32768

// ---------------- graph construction ----------------

__global__ void k_embed(const float* __restrict__ emb, const float* __restrict__ lw,
                        const float* __restrict__ lb, __bf16* __restrict__ nb) {
  int i = blockIdx.x;
  int k = threadIdx.x;
  float r = 0.0f;
  if (k < NDIM) {
    float acc = lb[k];
    for (int j = 0; j < NDIM; ++j) acc += emb[i * NDIM + j] * lw[k * NDIM + j];
    r = tanhf(3.0f * acc);
  }
  nb[i * KPAD + k] = (__bf16)r;
}

// C[i][j] = sum_k n1[i][k]*n2[j][k]  (1024x1024, K=64 padded), bf16 WMMA
__global__ void k_scores_wmma(const __bf16* __restrict__ n1b, const __bf16* __restrict__ n2b,
                              float* __restrict__ C) {
  int lane = threadIdx.x & 31;
  int tile = blockIdx.x;                 // 64*64 tiles of 16x16
  int tm = tile >> 6, tn = tile & 63;
  int half = lane >> 4, l = lane & 15;
  v8f c = {};
#pragma unroll
  for (int kk = 0; kk < KPAD; kk += 32) {
    v16bf a, b;
    int arow = tm * 16 + l;
    int bcol = tn * 16 + l;
#pragma unroll
    for (int v = 0; v < 8; ++v) {
      int ka = kk + ((v < 4) ? 0 : 16) + half * 8 + 2 * (v & 3);
      a[2 * v]     = n1b[arow * KPAD + ka];
      a[2 * v + 1] = n1b[arow * KPAD + ka + 1];
      int kb = kk + ((v < 4) ? 0 : 16) + half * 8 + 2 * (v & 3);
      b[2 * v]     = n2b[bcol * KPAD + kb];
      b[2 * v + 1] = n2b[bcol * KPAD + kb + 1];
    }
    c = __builtin_amdgcn_wmma_f32_16x16x32_bf16(false, a, false, b, (short)0, c, false, false);
  }
#pragma unroll
  for (int r = 0; r < 8; ++r) {
    int row = tm * 16 + half * 8 + r;
    int col = tn * 16 + l;
    C[row * NN + col] = c[r];
  }
}

// adj = relu(tanh(3*(C - C^T))); per-row top-20 mask -> dense masked A
__global__ void k_adj_topk(const float* __restrict__ C, float* __restrict__ A) {
  __shared__ float rowv[NN];
  __shared__ float rmax[256];
  __shared__ int   ridx[256];
  int i = blockIdx.x;
  int tid = threadIdx.x;
  for (int j = tid; j < NN; j += 256) {
    float a = C[i * NN + j] - C[j * NN + i];
    rowv[j] = fmaxf(tanhf(3.0f * a), 0.0f);
    A[i * NN + j] = 0.0f;
  }
  __syncthreads();
  for (int it = 0; it < 20; ++it) {
    float m = -1.0f; int mi = NN;
    for (int j = tid; j < NN; j += 256) {
      float v = rowv[j];
      if (v > m) { m = v; mi = j; }
    }
    rmax[tid] = m; ridx[tid] = mi;
    __syncthreads();
    for (int s = 128; s > 0; s >>= 1) {
      if (tid < s) {
        float vo = rmax[tid + s]; int io = ridx[tid + s];
        if (vo > rmax[tid] || (vo == rmax[tid] && io < ridx[tid])) { rmax[tid] = vo; ridx[tid] = io; }
      }
      __syncthreads();
    }
    if (tid == 0) {
      int j = ridx[0];
      A[i * NN + j] = rowv[j];
      rowv[j] = -2.0f;
    }
    __syncthreads();
  }
}

// fixed-degree normalized rows of (A+I) or (A^T+I)
__global__ void k_build_csr(const float* __restrict__ A, int transposed,
                            int* __restrict__ idx, float* __restrict__ val) {
  int v = blockIdx.x * blockDim.x + threadIdx.x;
  if (v >= NN) return;
  float rowsum = 1.0f;
  for (int w = 0; w < NN; ++w) {
    float a = transposed ? A[w * NN + v] : A[v * NN + w];
    if (a > 0.0f) rowsum += a;
  }
  float inv = 1.0f / rowsum;
  int c = 0;
  idx[v * NNZ + c] = v; val[v * NNZ + c] = inv; ++c;
  for (int w = 0; w < NN && c < NNZ; ++w) {
    float a = transposed ? A[w * NN + v] : A[v * NN + w];
    if (a > 0.0f) { idx[v * NNZ + c] = w; val[v * NNZ + c] = a * inv; ++c; }
  }
  for (; c < NNZ; ++c) { idx[v * NNZ + c] = v; val[v * NNZ + c] = 0.0f; }
}

// ---------------- front end ----------------

// start 1x1 conv on left-zero-padded input -> X [B,32,N,19]
__global__ void k_start(const float* __restrict__ in, const float* __restrict__ w,
                        const float* __restrict__ b, float* __restrict__ X) {
  int gid = blockIdx.x * blockDim.x + threadIdx.x;
  const int total = NB * CCH * NN * 19;
  if (gid >= total) return;
  int t = gid % 19;
  int n = (gid / 19) % NN;
  int co = (gid / (19 * NN)) % CCH;
  int bb = gid / (19 * NN * CCH);
  float acc = b[co];
  if (t >= 7) {
    int ti = t - 7;
    acc += w[co * 2 + 0] * in[((bb * 2 + 0) * NN + n) * 12 + ti]
         + w[co * 2 + 1] * in[((bb * 2 + 1) * NN + n) * 12 + ti];
  }
  X[gid] = acc;
}

// skip0: width-19 conv over padded input -> skip [B,64,N]
__global__ void k_skip0(const float* __restrict__ in, const float* __restrict__ w,
                        const float* __restrict__ b, float* __restrict__ skip) {
  int gid = blockIdx.x * blockDim.x + threadIdx.x;
  const int total = NB * SKIPCH * NN;
  if (gid >= total) return;
  int n = gid % NN;
  int s = (gid / NN) % SKIPCH;
  int bb = gid / (NN * SKIPCH);
  float acc = b[s];
  for (int ci = 0; ci < 2; ++ci)
    for (int t = 7; t < 19; ++t)
      acc += w[(s * 2 + ci) * 19 + t] * in[((bb * 2 + ci) * NN + n) * 12 + (t - 7)];
  skip[gid] = acc;
}

// ---------------- per-layer kernels ----------------

struct IncW {
  const float* fw[4]; const float* fb[4];
  const float* gw[4]; const float* gb[4];
};

__global__ void k_inception(const float* __restrict__ X, float* __restrict__ G,
                            IncW p, int Tin, int Tout) {
  int gid = blockIdx.x * blockDim.x + threadIdx.x;
  int total = NB * CCH * NN * Tout;
  if (gid >= total) return;
  int t = gid % Tout;
  int n = (gid / Tout) % NN;
  int c = (gid / (Tout * NN)) % CCH;
  int bb = gid / (Tout * NN * CCH);
  int br = c >> 3, oc = c & 7;
  int k = (br == 0) ? 2 : (br == 1) ? 3 : (br == 2) ? 6 : 7;
  int off = 7 - k;
  const float* fw = p.fw[br];
  const float* gw = p.gw[br];
  float F = p.fb[br][oc];
  float Gv = p.gb[br][oc];
  for (int ci = 0; ci < CCH; ++ci) {
    const float* xp = &X[((bb * CCH + ci) * NN + n) * Tin + t + off];
    const float* fwp = &fw[(oc * CCH + ci) * k];
    const float* gwp = &gw[(oc * CCH + ci) * k];
    for (int kk = 0; kk < k; ++kk) {
      float xv = xp[kk];
      F += fwp[kk] * xv;
      Gv += gwp[kk] * xv;
    }
  }
  G[gid] = tanhf(F) * (1.0f / (1.0f + expf(-Gv)));
}

__global__ void k_skipconv(const float* __restrict__ G, const float* __restrict__ sw,
                           const float* __restrict__ sb, float* __restrict__ skip, int Tk) {
  int gid = blockIdx.x * blockDim.x + threadIdx.x;
  const int total = NB * SKIPCH * NN;
  if (gid >= total) return;
  int n = gid % NN;
  int s = (gid / NN) % SKIPCH;
  int bb = gid / (NN * SKIPCH);
  float acc = sb[s];
  for (int ci = 0; ci < CCH; ++ci)
    for (int t = 0; t < Tk; ++t)
      acc += sw[(s * CCH + ci) * Tk + t] * G[((bb * CCH + ci) * NN + n) * Tk + t];
  skip[gid] += acc;
}

__global__ void k_mixinit(float* __restrict__ OUT, const float* __restrict__ g1b,
                          const float* __restrict__ g2b, int T) {
  int gid = blockIdx.x * blockDim.x + threadIdx.x;
  int total = NB * CCH * NN * T;
  if (gid >= total) return;
  int co = (gid / (T * NN)) % CCH;
  OUT[gid] = g1b[co] + g2b[co];
}

// OUT += W_hop @ H (1x1 over channels); W is [32,96], hop slice of 32
__global__ void k_mixaccum(float* __restrict__ OUT, const float* __restrict__ H,
                           const float* __restrict__ W, int hop, int T) {
  int gid = blockIdx.x * blockDim.x + threadIdx.x;
  int total = NB * CCH * NN * T;
  if (gid >= total) return;
  int t = gid % T;
  int n = (gid / T) % NN;
  int co = (gid / (T * NN)) % CCH;
  int bb = gid / (T * NN * CCH);
  float acc = 0.0f;
  const float* wr = &W[co * 96 + hop * CCH];
  for (int ci = 0; ci < CCH; ++ci)
    acc += wr[ci] * H[((bb * CCH + ci) * NN + n) * T + t];
  OUT[gid] += acc;
}

// Hout = alpha*X0 + (1-alpha)*A_sparse @ Hin  (gather SpMM, <=24 nnz/row)
__global__ void k_prop(const float* __restrict__ Hin, const float* __restrict__ X0,
                       float* __restrict__ Hout, const int* __restrict__ idx,
                       const float* __restrict__ val, int T) {
  int gid = blockIdx.x * blockDim.x + threadIdx.x;
  int total = NB * CCH * NN * T;
  if (gid >= total) return;
  int t = gid % T;
  int v = (gid / T) % NN;
  int c = (gid / (T * NN)) % CCH;
  int bb = gid / (T * NN * CCH);
  float acc = 0.0f;
#pragma unroll 4
  for (int e = 0; e < NNZ; ++e) {
    float a = val[v * NNZ + e];
    int w = idx[v * NNZ + e];
    acc += a * Hin[((bb * CCH + c) * NN + w) * T + t];
  }
  Hout[gid] = 0.05f * X0[gid] + 0.95f * acc;
}

__global__ void k_zero(float* __restrict__ p, int n) {
  int i = blockIdx.x * blockDim.x + threadIdx.x;
  if (i < n) p[i] = 0.0f;
}

// OUT += residual tail of X; accumulate per-batch sum/sumsq
__global__ void k_resid_stats(float* __restrict__ OUT, const float* __restrict__ X,
                              float* __restrict__ stats, int Tin, int Tout) {
  int bb = blockIdx.y;
  int per = CCH * NN * Tout;
  int i = blockIdx.x * blockDim.x + threadIdx.x;
  int tid = threadIdx.x;
  float r = 0.0f;
  if (i < per) {
    int t = i % Tout;
    int n = (i / Tout) % NN;
    int c = i / (Tout * NN);
    int gi = ((bb * CCH + c) * NN + n) * Tout + t;
    r = OUT[gi] + X[((bb * CCH + c) * NN + n) * Tin + t + (Tin - Tout)];
    OUT[gi] = r;
  }
  __shared__ float s1[256];
  __shared__ float s2[256];
  s1[tid] = r; s2[tid] = r * r;
  __syncthreads();
  for (int s = 128; s > 0; s >>= 1) {
    if (tid < s) { s1[tid] += s1[tid + s]; s2[tid] += s2[tid + s]; }
    __syncthreads();
  }
  if (tid == 0) {
    atomicAdd(&stats[bb * 2 + 0], s1[0]);
    atomicAdd(&stats[bb * 2 + 1], s2[0]);
  }
}

__global__ void k_ln(const float* __restrict__ OUT, const float* __restrict__ stats,
                     const float* __restrict__ nw, const float* __restrict__ nb,
                     float* __restrict__ X, int T) {
  int gid = blockIdx.x * blockDim.x + threadIdx.x;
  int total = NB * CCH * NN * T;
  if (gid >= total) return;
  int t = gid % T;
  int n = (gid / T) % NN;
  int c = (gid / (T * NN)) % CCH;
  int bb = gid / (T * NN * CCH);
  float cnt = (float)(CCH * NN * T);
  float mean = stats[bb * 2 + 0] / cnt;
  float var = stats[bb * 2 + 1] / cnt - mean * mean;
  float rstd = rsqrtf(var + 1e-5f);
  int a = (c * NN + n) * T + t;
  X[gid] = (OUT[gid] - mean) * rstd * nw[a] + nb[a];
}

// ---------------- back end ----------------

// skip += skipE(X); relu; pack bf16 as [64][32768] (k-major for WMMA B)
__global__ void k_skipE(const float* __restrict__ X, const float* __restrict__ w,
                        const float* __restrict__ b, const float* __restrict__ skip,
                        __bf16* __restrict__ sb16) {
  int gid = blockIdx.x * blockDim.x + threadIdx.x;
  const int total = NB * SKIPCH * NN;
  if (gid >= total) return;
  int n = gid % NN;
  int s = (gid / NN) % SKIPCH;
  int bb = gid / (NN * SKIPCH);
  float acc = b[s];
  for (int ci = 0; ci < CCH; ++ci)
    acc += w[s * CCH + ci] * X[(bb * CCH + ci) * NN + n];  // T==1
  float r = fmaxf(skip[gid] + acc, 0.0f);
  sb16[s * COLS + (bb * NN + n)] = (__bf16)r;
}

__global__ void k_cvt_w(const float* __restrict__ e1w, const float* __restrict__ e2w,
                        __bf16* __restrict__ e1wb, __bf16* __restrict__ e2wb) {
  int i = blockIdx.x * blockDim.x + threadIdx.x;
  if (i < 128 * 64) e1wb[i] = (__bf16)e1w[i];
  if (i < 16 * 128) {
    int o = i / 128, k = i % 128;
    e2wb[i] = (o < 12) ? (__bf16)e2w[o * 128 + k] : (__bf16)0.0f;
  }
}

// Y1 = relu(e1_w @ srelu + e1_b)  : [128 x 32768], K=64, bf16 WMMA
__global__ void k_e1_wmma(const __bf16* __restrict__ w, const __bf16* __restrict__ Bm,
                          const float* __restrict__ bias, __bf16* __restrict__ Y) {
  int lane = threadIdx.x & 31;
  int tile = blockIdx.x;                       // 8 * 2048
  int tm = tile / 2048, tn = tile % 2048;
  int half = lane >> 4, l = lane & 15;
  v8f c = {};
#pragma unroll
  for (int kk = 0; kk < 64; kk += 32) {
    v16bf a, b;
    int arow = tm * 16 + l;
    int bcol = tn * 16 + l;
#pragma unroll
    for (int v = 0; v < 8; ++v) {
      int ka = kk + ((v < 4) ? 0 : 16) + half * 8 + 2 * (v & 3);
      a[2 * v]     = w[arow * 64 + ka];
      a[2 * v + 1] = w[arow * 64 + ka + 1];
      int kb = kk + ((v < 4) ? 0 : 16) + half * 8 + 2 * (v & 3);
      b[2 * v]     = Bm[kb * COLS + bcol];
      b[2 * v + 1] = Bm[(kb + 1) * COLS + bcol];
    }
    c = __builtin_amdgcn_wmma_f32_16x16x32_bf16(false, a, false, b, (short)0, c, false, false);
  }
#pragma unroll
  for (int r = 0; r < 8; ++r) {
    int row = tm * 16 + half * 8 + r;
    int col = tn * 16 + l;
    float vv = fmaxf(c[r] + bias[row], 0.0f);
    Y[row * COLS + col] = (__bf16)vv;
  }
}

// out = e2_w @ Y1 + e2_b : [12(pad16) x 32768], K=128, bf16 WMMA
__global__ void k_e2_wmma(const __bf16* __restrict__ w, const __bf16* __restrict__ Bm,
                          const float* __restrict__ bias, float* __restrict__ out) {
  int lane = threadIdx.x & 31;
  int tn = blockIdx.x;                         // 2048 column tiles
  int half = lane >> 4, l = lane & 15;
  v8f c = {};
#pragma unroll
  for (int kk = 0; kk < 128; kk += 32) {
    v16bf a, b;
    int bcol = tn * 16 + l;
#pragma unroll
    for (int v = 0; v < 8; ++v) {
      int ka = kk + ((v < 4) ? 0 : 16) + half * 8 + 2 * (v & 3);
      a[2 * v]     = w[l * 128 + ka];
      a[2 * v + 1] = w[l * 128 + ka + 1];
      int kb = kk + ((v < 4) ? 0 : 16) + half * 8 + 2 * (v & 3);
      b[2 * v]     = Bm[kb * COLS + bcol];
      b[2 * v + 1] = Bm[(kb + 1) * COLS + bcol];
    }
    c = __builtin_amdgcn_wmma_f32_16x16x32_bf16(false, a, false, b, (short)0, c, false, false);
  }
#pragma unroll
  for (int r = 0; r < 8; ++r) {
    int o = half * 8 + r;
    if (o < 12) {
      int col = tn * 16 + l;
      int bb = col >> 10, n = col & 1023;
      out[(bb * 12 + o) * NN + n] = c[r] + bias[o];
    }
  }
}

// ---------------- launcher ----------------

extern "C" void kernel_launch(void* const* d_in, const int* in_sizes, int n_in,
                              void* d_out, int out_size, void* d_ws, size_t ws_size,
                              hipStream_t stream) {
  (void)in_sizes; (void)n_in; (void)out_size; (void)ws_size;
  auto F = [&](int i) { return (const float*)d_in[i]; };

  const float* input  = F(0);
  const float* startw = F(1);  const float* startb = F(2);
  const float* skip0w = F(3);  const float* skip0b = F(4);
  const float* emb1 = F(5);    const float* emb2 = F(6);
  const float* lin1w = F(7);   const float* lin1b = F(8);
  const float* lin2w = F(9);   const float* lin2b = F(10);
  // 11/12: skipE_w/b  13/14: e1_w/b  15/16: e2_w/b ; layers at 17 + 24*L

  char* wsp = (char*)d_ws;
  size_t off = 0;
  auto carve = [&](size_t bytes) -> char* {
    char* p = wsp + off;
    off += (bytes + 255) & ~(size_t)255;
    return p;
  };
  float*  Cs   = (float*)carve((size_t)NN * NN * 4);
  float*  Ad   = (float*)carve((size_t)NN * NN * 4);
  __bf16* n1b  = (__bf16*)carve((size_t)NN * KPAD * 2);
  __bf16* n2b  = (__bf16*)carve((size_t)NN * KPAD * 2);
  int*    idx1 = (int*)carve((size_t)NN * NNZ * 4);
  float*  val1 = (float*)carve((size_t)NN * NNZ * 4);
  int*    idx2 = (int*)carve((size_t)NN * NNZ * 4);
  float*  val2 = (float*)carve((size_t)NN * NNZ * 4);
  float*  X    = (float*)carve((size_t)NB * CCH * NN * 19 * 4);
  float*  G    = (float*)carve((size_t)NB * CCH * NN * 13 * 4);
  float*  H    = (float*)carve((size_t)NB * CCH * NN * 13 * 4);
  float*  H2   = (float*)carve((size_t)NB * CCH * NN * 13 * 4);
  float*  OUT  = (float*)carve((size_t)NB * CCH * NN * 13 * 4);
  float*  skip = (float*)carve((size_t)NB * SKIPCH * NN * 4);
  float*  stats = (float*)carve(64 * 4);
  __bf16* sb16 = (__bf16*)carve((size_t)SKIPCH * COLS * 2);
  __bf16* y1b  = (__bf16*)carve((size_t)128 * COLS * 2);
  __bf16* e1wb = (__bf16*)carve((size_t)128 * 64 * 2);
  __bf16* e2wb = (__bf16*)carve((size_t)16 * 128 * 2);

  // graph construction
  k_embed<<<NN, 64, 0, stream>>>(emb1, lin1w, lin1b, n1b);
  k_embed<<<NN, 64, 0, stream>>>(emb2, lin2w, lin2b, n2b);
  k_scores_wmma<<<64 * 64, 32, 0, stream>>>(n1b, n2b, Cs);
  k_adj_topk<<<NN, 256, 0, stream>>>(Cs, Ad);
  k_build_csr<<<4, 256, 0, stream>>>(Ad, 0, idx1, val1);
  k_build_csr<<<4, 256, 0, stream>>>(Ad, 1, idx2, val2);

  // front end
  {
    int tot = NB * CCH * NN * 19;
    k_start<<<(tot + 255) / 256, 256, 0, stream>>>(input, startw, startb, X);
    tot = NB * SKIPCH * NN;
    k_skip0<<<(tot + 255) / 256, 256, 0, stream>>>(input, skip0w, skip0b, skip);
  }

  const int Tins[3]  = {19, 13, 7};
  const int Touts[3] = {13, 7, 1};
  for (int L = 0; L < 3; ++L) {
    int base = 17 + L * 24;
    IncW p;
    for (int i = 0; i < 4; ++i) {
      p.fw[i] = F(base + i);      p.fb[i] = F(base + 4 + i);
      p.gw[i] = F(base + 8 + i);  p.gb[i] = F(base + 12 + i);
    }
    const float* sw  = F(base + 16); const float* sb  = F(base + 17);
    const float* g1w = F(base + 18); const float* g1b = F(base + 19);
    const float* g2w = F(base + 20); const float* g2b = F(base + 21);
    const float* nw  = F(base + 22); const float* nbv = F(base + 23);
    int Tin = Tins[L], Tout = Touts[L];
    int tot = NB * CCH * NN * Tout;
    int grd = (tot + 255) / 256;

    k_inception<<<grd, 256, 0, stream>>>(X, G, p, Tin, Tout);
    {
      int t2 = NB * SKIPCH * NN;
      k_skipconv<<<(t2 + 255) / 256, 256, 0, stream>>>(G, sw, sb, skip, Tout);
    }
    k_mixinit<<<grd, 256, 0, stream>>>(OUT, g1b, g2b, Tout);
    // direction 1 (adp)
    k_mixaccum<<<grd, 256, 0, stream>>>(OUT, G, g1w, 0, Tout);
    k_prop<<<grd, 256, 0, stream>>>(G, G, H, idx1, val1, Tout);
    k_mixaccum<<<grd, 256, 0, stream>>>(OUT, H, g1w, 1, Tout);
    k_prop<<<grd, 256, 0, stream>>>(H, G, H2, idx1, val1, Tout);
    k_mixaccum<<<grd, 256, 0, stream>>>(OUT, H2, g1w, 2, Tout);
    // direction 2 (adp.T)
    k_mixaccum<<<grd, 256, 0, stream>>>(OUT, G, g2w, 0, Tout);
    k_prop<<<grd, 256, 0, stream>>>(G, G, H, idx2, val2, Tout);
    k_mixaccum<<<grd, 256, 0, stream>>>(OUT, H, g2w, 1, Tout);
    k_prop<<<grd, 256, 0, stream>>>(H, G, H2, idx2, val2, Tout);
    k_mixaccum<<<grd, 256, 0, stream>>>(OUT, H2, g2w, 2, Tout);
    // residual + LayerNorm
    k_zero<<<1, 64, 0, stream>>>(stats, 64);
    {
      int per = CCH * NN * Tout;
      dim3 gr((per + 255) / 256, NB);
      k_resid_stats<<<gr, 256, 0, stream>>>(OUT, X, stats, Tin, Tout);
    }
    k_ln<<<grd, 256, 0, stream>>>(OUT, stats, nw, nbv, X, Tout);
  }

  // back end
  {
    int t2 = NB * SKIPCH * NN;
    k_skipE<<<(t2 + 255) / 256, 256, 0, stream>>>(X, F(11), F(12), skip, sb16);
  }
  k_cvt_w<<<(8192 + 255) / 256, 256, 0, stream>>>(F(13), F(15), e1wb, e2wb);
  k_e1_wmma<<<8 * 2048, 32, 0, stream>>>(e1wb, sb16, F(14), y1b);
  k_e2_wmma<<<2048, 32, 0, stream>>>(e2wb, y1b, F(16), (float*)d_out);
}
